// FusionMultiHeadAttention_49538152792908
// MI455X (gfx1250) — matrix operand order repaired and checked by reading, not compiled, not run on hardware
//
#include <hip/hip_runtime.h>
#include <hip/hip_bf16.h>

typedef __attribute__((ext_vector_type(16))) _Float16 v16h;
typedef __attribute__((ext_vector_type(8)))  _Float16 v8h;
typedef __attribute__((ext_vector_type(2)))  _Float16 h2;
typedef __attribute__((ext_vector_type(2)))  __fp16   fp16x2;   // cvt_pkrtz native type
typedef __attribute__((ext_vector_type(8)))  float    v8f;
typedef __attribute__((ext_vector_type(4)))  unsigned int v4u;

union FragU { v16h h; v4u u[2]; };
union H8U   { v4u u; v8h h; };
union PkU   { fp16x2 v; unsigned int u; };

__device__ __forceinline__ v8f wmma_f16(v16h a, v16h b, v8f c) {
    return __builtin_amdgcn_wmma_f32_16x16x32_f16(false, a, false, b, (short)0, c, false, false);
}

__device__ __forceinline__ v8f v8f_zero() {
    v8f z;
    #pragma unroll
    for (int i = 0; i < 8; ++i) z[i] = 0.0f;
    return z;
}

// Pack a float4 into 4 f16 (2x v_cvt_pk) and store as one ds_store_b64.
__device__ __forceinline__ void store_pk4(_Float16* dst, float4 f) {
    PkU lo, hi;
    lo.v = __builtin_amdgcn_cvt_pkrtz(f.x, f.y);
    hi.v = __builtin_amdgcn_cvt_pkrtz(f.z, f.w);
    uint2 w; w.x = lo.u; w.y = hi.u;
    *(uint2*)dst = w;
}

// CDNA5 async global->LDS copy (16B per lane), tracked by ASYNCcnt.
// LDS operand = low 32 bits of the flat shared-pointer (ISA 10.2: LDS aperture
// truncates flat addr to addr[31:0]).
__device__ __forceinline__ void async_copy_b128(const _Float16* g, _Float16* l) {
    unsigned lds = (unsigned)(unsigned long long)(uintptr_t)l;
    unsigned long long ga = (unsigned long long)(uintptr_t)g;
    asm volatile("global_load_async_to_lds_b128 %0, %1, off" :: "v"(lds), "v"(ga) : "memory");
}
__device__ __forceinline__ void wait_async0() {
#if __has_builtin(__builtin_amdgcn_s_wait_asynccnt)
    __builtin_amdgcn_s_wait_asynccnt(0);
#else
    asm volatile("s_wait_asynccnt 0" ::: "memory");
#endif
}

// ---------------------------------------------------------------------------
// C[M,NN] = A[M,KD] @ W[NN,KD]^T   (torch Linear, no bias)
// A: f32 (packed-converted to f16 while staging) or f16; C: f16 or f32.
// 256 threads = 8 waves; tile 128x128x32; wave grid 4x2 (32 rows x 64 cols).
// Double-buffered LDS + register prefetch: one barrier per K-step, global
// loads in flight while WMMAs run on the other buffer.
// ---------------------------------------------------------------------------
template <bool A_F32, bool OUT_F32>
__global__ __launch_bounds__(256)
void gemm_nt_kernel(const void* __restrict__ Ap, const float* __restrict__ Wp,
                    void* __restrict__ Cp, int M, int NN, int KD)
{
    constexpr int BM = 128, BN = 128, BK = 32;
    constexpr int LDP = 40;                      // padded stride (halves): 80B, 16B-aligned, conflict-free
    __shared__ __align__(16) _Float16 ldsA[2][BM * LDP];
    __shared__ __align__(16) _Float16 ldsB[2][BN * LDP];

    const int tid  = threadIdx.x;
    const int lane = tid & 31;
    const int wid  = tid >> 5;
    const int wm   = wid >> 1;       // 0..3 : 32-row slab
    const int wn   = wid & 1;        // 0..1 : 64-col slab
    const int ll   = lane & 15;
    const int lg   = lane >> 4;
    const int mBase = blockIdx.y * BM;
    const int nBase = blockIdx.x * BN;

    float4 ra[4];   // A prefetch (f32 path)
    v4u    ra16[2]; // A prefetch (f16 path)
    float4 rb[4];   // W prefetch

    auto gload = [&](int k0) {
        if (A_F32) {
            const float* A = (const float*)Ap;
            #pragma unroll
            for (int i = 0; i < 4; ++i) {
                int idx = tid + i * 256, row = idx >> 3, kq = (idx & 7) << 2;
                ra[i] = *(const float4*)(A + (size_t)(mBase + row) * KD + k0 + kq);
            }
        } else {
            const _Float16* A = (const _Float16*)Ap;
            #pragma unroll
            for (int i = 0; i < 2; ++i) {
                int idx = tid + i * 256, row = idx >> 2, kq = (idx & 3) << 3;
                ra16[i] = *(const v4u*)(A + (size_t)(mBase + row) * KD + k0 + kq);
            }
        }
        #pragma unroll
        for (int i = 0; i < 4; ++i) {
            int idx = tid + i * 256, row = idx >> 3, kq = (idx & 7) << 2;
            rb[i] = *(const float4*)(Wp + (size_t)(nBase + row) * KD + k0 + kq);
        }
    };
    auto stage = [&](int buf) {
        if (A_F32) {
            #pragma unroll
            for (int i = 0; i < 4; ++i) {
                int idx = tid + i * 256, row = idx >> 3, kq = (idx & 7) << 2;
                store_pk4(&ldsA[buf][row * LDP + kq], ra[i]);
            }
        } else {
            #pragma unroll
            for (int i = 0; i < 2; ++i) {
                int idx = tid + i * 256, row = idx >> 2, kq = (idx & 3) << 3;
                *(v4u*)&ldsA[buf][row * LDP + kq] = ra16[i];
            }
        }
        #pragma unroll
        for (int i = 0; i < 4; ++i) {
            int idx = tid + i * 256, row = idx >> 3, kq = (idx & 7) << 2;
            store_pk4(&ldsB[buf][row * LDP + kq], rb[i]);
        }
    };

    v8f acc[2][4];
    #pragma unroll
    for (int mt = 0; mt < 2; ++mt)
        #pragma unroll
        for (int nt = 0; nt < 4; ++nt) acc[mt][nt] = v8f_zero();

    const int NK = KD / BK;
    gload(0);
    stage(0);
    for (int kt = 0; kt < NK; ++kt) {
        __syncthreads();                         // buf[kt&1] visible to all waves
        if (kt + 1 < NK) gload((kt + 1) * BK);   // next tile's vmem in flight
        const int buf = kt & 1;

        FragU afr[2], bfr[4];
        #pragma unroll
        for (int mt = 0; mt < 2; ++mt) {
            int row = wm * 32 + mt * 16 + ll, ko = lg * 8;
            afr[mt].u[0] = *(const v4u*)&ldsA[buf][row * LDP + ko];
            afr[mt].u[1] = *(const v4u*)&ldsA[buf][row * LDP + ko + 16];
        }
        #pragma unroll
        for (int nt = 0; nt < 4; ++nt) {
            int row = wn * 64 + nt * 16 + ll, ko = lg * 8;
            bfr[nt].u[0] = *(const v4u*)&ldsB[buf][row * LDP + ko];
            bfr[nt].u[1] = *(const v4u*)&ldsB[buf][row * LDP + ko + 16];
        }
        #pragma unroll
        for (int mt = 0; mt < 2; ++mt)
            #pragma unroll
            for (int nt = 0; nt < 4; ++nt)
                acc[mt][nt] = wmma_f16(afr[mt].h, bfr[nt].h, acc[mt][nt]);

        if (kt + 1 < NK) stage((kt + 1) & 1);    // other buffer: no barrier needed
    }

    #pragma unroll
    for (int mt = 0; mt < 2; ++mt)
        #pragma unroll
        for (int nt = 0; nt < 4; ++nt)
            #pragma unroll
            for (int v = 0; v < 8; ++v) {
                int row = mBase + wm * 32 + mt * 16 + v + lg * 8;
                int col = nBase + wn * 64 + nt * 16 + ll;
                if (OUT_F32)
                    ((float*)Cp)[(size_t)row * NN + col] = acc[mt][nt][v];
                else
                    ((_Float16*)Cp)[(size_t)row * NN + col] = (_Float16)acc[mt][nt][v];
            }
}

// ---------------------------------------------------------------------------
// Flash attention, "+1" partition: attn = exp(s-m) / (1 + sum exp(s-m)).
// Grid (B*H, N/64); 128 threads = 4 waves x 16 query rows. 64-key chunks:
// 8 WMMA for S, 8 WMMA for PV per chunk. K tiles staged with CDNA5 async
// global->LDS copies (ASYNCcnt); V tiles transposed through registers with
// packed h2 stores. 1/32 scale pre-folded into Q frags. Double-buffered LDS.
// ---------------------------------------------------------------------------
__global__ __launch_bounds__(128)
void attn_kernel(const _Float16* __restrict__ Q, const _Float16* __restrict__ K,
                 const _Float16* __restrict__ V, _Float16* __restrict__ O,
                 int Nn, int Hn)
{
    constexpr int HD = 64, QB = 64, KB = 64;
    constexpr int LDK = 72;   // [key][hd] stride (halves): 144B, 16B-aligned
    constexpr int LDV = 72;   // [hd][key] stride
    constexpr int LDPW = 72;  // per-wave P stride
    __shared__ __align__(16) _Float16 ldsK[2][KB * LDK];
    __shared__ __align__(16) _Float16 ldsVt[2][HD * LDV];
    __shared__ __align__(16) _Float16 ldsP[4][16 * LDPW];

    const int b = blockIdx.x / Hn, h = blockIdx.x % Hn;
    const int qblk = blockIdx.y;
    const int tid = threadIdx.x, lane = tid & 31, wid = tid >> 5;
    const int ll = lane & 15, lg = lane >> 4;
    const size_t rowStride = (size_t)Hn * HD;
    const size_t base = (size_t)b * Nn * rowStride + (size_t)h * HD;

    // Q fragments (HD=64 -> two k=32 A-frags), pre-scaled by 1/sqrt(D_K)=1/32 (exact in f16)
    FragU qf[2];
    {
        const _Float16* qp = Q + base + (size_t)(qblk * QB + wid * 16 + ll) * rowStride;
        int ko = lg * 8;
        qf[0].u[0] = *(const v4u*)(qp + ko);
        qf[0].u[1] = *(const v4u*)(qp + ko + 16);
        qf[1].u[0] = *(const v4u*)(qp + 32 + ko);
        qf[1].u[1] = *(const v4u*)(qp + 32 + ko + 16);
        qf[0].h = qf[0].h * (_Float16)0.03125f;
        qf[1].h = qf[1].h * (_Float16)0.03125f;
    }

    // K tiles: async global->LDS, no VGPR round trip.
    auto issueK = [&](int kt, int buf) {
        #pragma unroll
        for (int i = 0; i < 4; ++i) {
            int idx = tid + i * 128, kk = idx >> 3, hq = (idx & 7) << 3;
            async_copy_b128(K + base + (size_t)(kt + kk) * rowStride + hq,
                            &ldsK[buf][kk * LDK + hq]);
        }
    };
    // V tiles: register prefetch + packed transpose stores.
    v4u rv0[2], rv1[2];
    auto gloadV = [&](int kt) {
        #pragma unroll
        for (int i = 0; i < 2; ++i) {                 // 2-key x 8-hd micro-tiles
            int idx = tid + i * 128, kp = idx >> 3, hq = (idx & 7) << 3;
            rv0[i] = *(const v4u*)(V + base + (size_t)(kt + 2 * kp)     * rowStride + hq);
            rv1[i] = *(const v4u*)(V + base + (size_t)(kt + 2 * kp + 1) * rowStride + hq);
        }
    };
    auto stageV = [&](int buf) {
        #pragma unroll
        for (int i = 0; i < 2; ++i) {
            int idx = tid + i * 128, kp = idx >> 3, hq = (idx & 7) << 3;
            H8U a, c; a.u = rv0[i]; c.u = rv1[i];
            #pragma unroll
            for (int e = 0; e < 8; ++e) {             // packed transpose stores
                h2 t; t.x = a.h[e]; t.y = c.h[e];
                *(h2*)&ldsVt[buf][(hq + e) * LDV + 2 * kp] = t;
            }
        }
    };

    float rowM[8], rowL[8];
    v8f accO[4];
    #pragma unroll
    for (int v = 0; v < 8; ++v) { rowM[v] = -1e30f; rowL[v] = 0.0f; }
    #pragma unroll
    for (int t = 0; t < 4; ++t) accO[t] = v8f_zero();

    const int NC = Nn / KB;
    issueK(0, 0);
    gloadV(0);
    stageV(0);
    for (int c2 = 0; c2 < NC; ++c2) {
        wait_async0();                 // asyncs targeting buf[c2&1] complete (wave-local)
        __syncthreads();               // ... and visible block-wide
        if (c2 + 1 < NC) {
            issueK((c2 + 1) * KB, (c2 + 1) & 1);   // async engine fills alt buffer
            gloadV((c2 + 1) * KB);                 // V vmem in flight during WMMAs
        }
        const int buf = c2 & 1;

        // ---- scores: S[16 q][64 keys] ----
        v8f accS[4];
        #pragma unroll
        for (int nt = 0; nt < 4; ++nt) accS[nt] = v8f_zero();
        #pragma unroll
        for (int nt = 0; nt < 4; ++nt) {
            #pragma unroll
            for (int c = 0; c < 2; ++c) {
                FragU bf;
                int key = nt * 16 + ll, ho = c * 32 + lg * 8;
                bf.u[0] = *(const v4u*)&ldsK[buf][key * LDK + ho];
                bf.u[1] = *(const v4u*)&ldsK[buf][key * LDK + ho + 16];
                accS[nt] = wmma_f16(qf[c].h, bf.h, accS[nt]);
            }
        }

        // ---- online softmax (row lives across a 16-lane half-wave) ----
        #pragma unroll
        for (int v = 0; v < 8; ++v) {
            float s0 = accS[0][v], s1 = accS[1][v], s2 = accS[2][v], s3 = accS[3][v];
            float m = fmaxf(fmaxf(s0, s1), fmaxf(s2, s3));
            #pragma unroll
            for (int msk = 1; msk < 16; msk <<= 1) m = fmaxf(m, __shfl_xor(m, msk, 32));
            float Mn = fmaxf(rowM[v], m);
            float corr = __expf(rowM[v] - Mn);
            float p0 = __expf(s0 - Mn), p1 = __expf(s1 - Mn);
            float p2 = __expf(s2 - Mn), p3 = __expf(s3 - Mn);
            float ps = (p0 + p1) + (p2 + p3);
            #pragma unroll
            for (int msk = 1; msk < 16; msk <<= 1) ps += __shfl_xor(ps, msk, 32);
            rowL[v] = rowL[v] * corr + ps;
            rowM[v] = Mn;
            #pragma unroll
            for (int t = 0; t < 4; ++t) accO[t][v] *= corr;
            int prow = v + lg * 8;
            ldsP[wid][prow * LDPW + ll]      = (_Float16)p0;
            ldsP[wid][prow * LDPW + ll + 16] = (_Float16)p1;
            ldsP[wid][prow * LDPW + ll + 32] = (_Float16)p2;
            ldsP[wid][prow * LDPW + ll + 48] = (_Float16)p3;
        }

        // ---- O += P @ Vc : two k=32 steps x 4 hd tiles ----
        #pragma unroll
        for (int c = 0; c < 2; ++c) {
            FragU pf;
            int ko = c * 32 + lg * 8;
            pf.u[0] = *(const v4u*)&ldsP[wid][ll * LDPW + ko];
            pf.u[1] = *(const v4u*)&ldsP[wid][ll * LDPW + ko + 16];
            #pragma unroll
            for (int t = 0; t < 4; ++t) {
                FragU bf;
                int hd = t * 16 + ll, vo = c * 32 + lg * 8;
                bf.u[0] = *(const v4u*)&ldsVt[buf][hd * LDV + vo];
                bf.u[1] = *(const v4u*)&ldsVt[buf][hd * LDV + vo + 16];
                accO[t] = wmma_f16(pf.h, bf.h, accO[t]);
            }
        }

        if (c2 + 1 < NC) stageV((c2 + 1) & 1);
    }

    // ---- finalize: divide by (1 + l) ----
    const int qrow0 = qblk * QB + wid * 16;
    #pragma unroll
    for (int v = 0; v < 8; ++v) {
        float r = 1.0f / (1.0f + rowL[v]);
        int row = qrow0 + v + lg * 8;
        #pragma unroll
        for (int t = 0; t < 4; ++t) {
            int hd = t * 16 + ll;
            O[base + (size_t)row * rowStride + hd] = (_Float16)(accO[t][v] * r);
        }
    }
}

// ---------------------------------------------------------------------------
extern "C" void kernel_launch(void* const* d_in, const int* in_sizes, int n_in,
                              void* d_out, int out_size, void* d_ws, size_t ws_size,
                              hipStream_t stream)
{
    (void)in_sizes; (void)n_in; (void)out_size; (void)ws_size;
    constexpr int B = 4, N = 2048, IN_DIM = 1024, D_K = 1024, H = 16;
    const int M = B * N;   // 8192

    const float* Xst = (const float*)d_in[0];
    const float* Xf  = (const float*)d_in[1];
    const float* Wq  = (const float*)d_in[2];
    const float* Wk  = (const float*)d_in[3];
    const float* Wv  = (const float*)d_in[4];
    const float* Wo  = (const float*)d_in[5];
    float* out = (float*)d_out;

    _Float16* Q16 = (_Float16*)d_ws;
    _Float16* K16 = Q16 + (size_t)M * D_K;
    _Float16* V16 = K16 + (size_t)M * D_K;
    _Float16* O16 = V16 + (size_t)M * D_K;

    dim3 gProj(D_K / 128, M / 128);
    gemm_nt_kernel<true, false><<<gProj, 256, 0, stream>>>(Xst, Wq, Q16, M, D_K, IN_DIM);
    gemm_nt_kernel<true, false><<<gProj, 256, 0, stream>>>(Xf,  Wk, K16, M, D_K, IN_DIM);
    gemm_nt_kernel<true, false><<<gProj, 256, 0, stream>>>(Xf,  Wv, V16, M, D_K, IN_DIM);

    attn_kernel<<<dim3(B * H, N / 64), 128, 0, stream>>>(Q16, K16, V16, O16, N, H);

    dim3 gOut(IN_DIM / 128, M / 128);
    gemm_nt_kernel<false, true><<<gOut, 256, 0, stream>>>(O16, Wo, out, M, IN_DIM, D_K);
}
